// PatchTSTBackbone_15204184228291
// MI455X (gfx1250) — matrix operand order repaired and checked by reading, not compiled, not run on hardware
//
#include <hip/hip_runtime.h>
#include <math.h>

// ---------------------------------------------------------------------------
// PatchTST backbone forward on MI455X (gfx1250): all GEMMs via
// v_wmma_f32_16x16x32_bf16 (bf16 operands, f32 accumulate); GEMM tiles are
// staged global->LDS by the Tensor Data Mover (tensor_load_to_lds) with
// hardware LDS padding, double-buffered, drained with s_wait_tensorcnt.
// B=64 S=4096 F=8 P=16 D=512 H=8 L=6 FF=2048 R=8 NP=256 DK=64
// ---------------------------------------------------------------------------

typedef __attribute__((ext_vector_type(16))) __bf16 v16bf;
typedef __attribute__((ext_vector_type(8)))  __bf16 v8bf;
typedef __attribute__((ext_vector_type(8)))  float  v8f;
typedef __attribute__((ext_vector_type(4)))  unsigned int u32x4;
typedef __attribute__((ext_vector_type(4)))  int i32x4;
typedef __attribute__((ext_vector_type(8)))  int i32x8;

#define GLD  72   // GEMM LDS row stride (bf16): 64 data + 8 pad = 144B, 16B aligned, conflict-free
#define PLDT 40   // attention P-staging stride: 32 data + 8 pad = 80B

#if defined(__HIP_DEVICE_COMPILE__) && __has_builtin(__builtin_amdgcn_tensor_load_to_lds)
#define HAS_TDM 1
#else
#define HAS_TDM 0
#endif

#if __has_include(<hip/amd_detail/amd_gfx1250_TDM.h>)
#define TDM_ARGS6 1
#else
#define TDM_ARGS6 0
#endif

__device__ __forceinline__ v16bf cat16(v8bf lo, v8bf hi) {
  return __builtin_shufflevector(lo, hi, 0,1,2,3,4,5,6,7,8,9,10,11,12,13,14,15);
}

__device__ __forceinline__ v8f vzero8() {
  v8f z = {0.f,0.f,0.f,0.f,0.f,0.f,0.f,0.f};
  return z;
}

__device__ __forceinline__ unsigned lds_off(const void* p) {
  // generic pointer to a __shared__ object: low 32 bits are the LDS byte offset
  return (unsigned)(unsigned long long)(uintptr_t)p;
}

#if HAS_TDM
// ---------------------------------------------------------------------------
// TDM: DMA a 2D tile of 16-bit elements (tile_d0 x tile_d1, row stride
// `stride_elems`) from global to LDS, inserting 4 DWORDs of padding every
// `row bytes` (pad codes passed in) so the LDS image has stride GLD.
// D# layout per CDNA5 ISA 8.3/8.4 (group0: count/lds_addr/global_addr/type,
// group1: data_size/pad/dims/strides). 2D tensor -> groups 2/3 zero.
// ---------------------------------------------------------------------------
__device__ __forceinline__ void tdm_load_tile_2d(
    const void* gaddr, unsigned lds_addr,
    unsigned tile_d0, unsigned tile_d1,
    unsigned tensor_d0, unsigned tensor_d1,
    unsigned stride_elems, unsigned pad_interval_code, unsigned pad_amount_code)
{
  unsigned long long ga = (unsigned long long)(uintptr_t)gaddr;
  u32x4 g0;
  g0[0] = 1u;                                   // count=1 (user descriptor)
  g0[1] = lds_addr;                             // LDS byte address
  g0[2] = (unsigned)ga;                         // global_addr[31:0]
  g0[3] = (unsigned)((ga >> 32) & 0x01FFFFFFu)  // global_addr[56:32]
        | (2u << 30);                           // type = 2 ("image")
  i32x8 g1;
  g1[0] = (int)((1u << 16)                      // data_size = 1 -> 2 bytes
        | (1u << 20)                            // pad_enable
        | (pad_interval_code << 22)             // DWORDs before pad
        | (pad_amount_code << 25));             // pad DWORDs
  g1[1] = (int)((tensor_d0 & 0xFFFFu) << 16);                            // dim0[15:0]
  g1[2] = (int)(((tensor_d0 >> 16) & 0xFFFFu) | ((tensor_d1 & 0xFFFFu) << 16));
  g1[3] = (int)(((tensor_d1 >> 16) & 0xFFFFu) | (tile_d0 << 16));        // tile_dim0
  g1[4] = (int)tile_d1;                         // tile_dim1 (tile_dim2 = 0)
  g1[5] = (int)stride_elems;                    // tensor_dim0_stride[31:0]
  g1[6] = 0;                                    // stride[47:32] | dim1_stride lo
  g1[7] = 0;
  i32x4 z4 = {0, 0, 0, 0};
#if TDM_ARGS6
  i32x8 z8 = {0, 0, 0, 0, 0, 0, 0, 0};
  __builtin_amdgcn_tensor_load_to_lds(g0, g1, z4, z4, z8, 0);
#else
  __builtin_amdgcn_tensor_load_to_lds(g0, g1, z4, z4, 0);
#endif
}
#endif  // HAS_TDM

// ---------------------------------------------------------------------------
// Tiled bf16 WMMA GEMM: C[M,N] = A[M,K] (row major bf16) * Bt[N,K] (N-major)
// Epilogues: 0 = +bias -> f32, 1 = +bias -> bf16,
//            2 = +bias + resid -> f32, 3 = gelu(+bias) -> bf16
// Block: 256 threads (8 waves), tile 128x128, wave computes 32x64.
// K step 64, double-buffered LDS; tiles staged by TDM (fallback: manual).
// ---------------------------------------------------------------------------
enum { EPI_F32 = 0, EPI_BF16 = 1, EPI_RESID_F32 = 2, EPI_GELU_BF16 = 3 };

template<int EPI>
__global__ __launch_bounds__(256)
void gemm_bf16(const __bf16* __restrict__ A, const __bf16* __restrict__ Bt,
               const float* __restrict__ bias, const float* __restrict__ resid,
               void* __restrict__ outp, int M, int N, int K)
{
  __shared__ __bf16 Ash[2][128 * GLD];
  __shared__ __bf16 Bsh[2][128 * GLD];

  const int t    = threadIdx.x;
  const int lane = t & 31;
  const int wid  = t >> 5;
  const int lm   = lane & 15;
  const int g    = lane >> 4;
  const int blockM = blockIdx.y * 128;
  const int blockN = blockIdx.x * 128;
  const int wm = (wid & 3) * 32;   // wave M offset within tile
  const int wn = (wid >> 2) * 64;  // wave N offset within tile

  v8f acc[2][4];
#pragma unroll
  for (int mi = 0; mi < 2; ++mi)
#pragma unroll
    for (int ni = 0; ni < 4; ++ni) acc[mi][ni] = vzero8();

  const int nk = K >> 6;  // K is a multiple of 64 for every call site

#if HAS_TDM
  // Row = 64 bf16 = 128B = 32 DWORDs -> pad_interval code 4 (32 DWORDs);
  // pad 8 bf16 = 16B = 4 DWORDs -> pad_amount code 3.
  if (wid == 0) {
    tdm_load_tile_2d(A + (size_t)blockM * K, lds_off(&Ash[0][0]),
                     64u, 128u, (unsigned)K, (unsigned)(M - blockM), (unsigned)K, 4u, 3u);
    tdm_load_tile_2d(Bt + (size_t)blockN * K, lds_off(&Bsh[0][0]),
                     64u, 128u, (unsigned)K, (unsigned)(N - blockN), (unsigned)K, 4u, 3u);
    __builtin_amdgcn_s_wait_tensorcnt(0);
  }
  __syncthreads();

  for (int i = 0; i < nk; ++i) {
    const int cur = i & 1;
    if (wid == 0 && i + 1 < nk) {   // issue next tiles; DMA overlaps compute
      int koff = (i + 1) << 6;
      tdm_load_tile_2d(A + (size_t)blockM * K + koff, lds_off(&Ash[cur ^ 1][0]),
                       64u, 128u, (unsigned)(K - koff), (unsigned)(M - blockM), (unsigned)K, 4u, 3u);
      tdm_load_tile_2d(Bt + (size_t)blockN * K + koff, lds_off(&Bsh[cur ^ 1][0]),
                       64u, 128u, (unsigned)(K - koff), (unsigned)(N - blockN), (unsigned)K, 4u, 3u);
    }
    const __bf16* As = &Ash[cur][0];
    const __bf16* Bs = &Bsh[cur][0];
#pragma unroll
    for (int k2 = 0; k2 < 64; k2 += 32) {
      v16bf af[2];
#pragma unroll
      for (int mi = 0; mi < 2; ++mi) {
        const __bf16* ap = As + (wm + mi * 16 + lm) * GLD + k2 + g * 8;
        af[mi] = cat16(*(const v8bf*)ap, *(const v8bf*)(ap + 16));
      }
      v16bf bfr[4];
#pragma unroll
      for (int ni = 0; ni < 4; ++ni) {
        const __bf16* bp = Bs + (wn + ni * 16 + lm) * GLD + k2 + g * 8;
        bfr[ni] = cat16(*(const v8bf*)bp, *(const v8bf*)(bp + 16));
      }
#pragma unroll
      for (int mi = 0; mi < 2; ++mi)
#pragma unroll
        for (int ni = 0; ni < 4; ++ni)
          acc[mi][ni] = __builtin_amdgcn_wmma_f32_16x16x32_bf16(
              false, af[mi], false, bfr[ni], (short)0, acc[mi][ni], false, false);
    }
    if (wid == 0 && i + 1 < nk) __builtin_amdgcn_s_wait_tensorcnt(0);
    __syncthreads();
  }
#else
  // Manual staging fallback (also what the host pass parses).
  const int lr  = t >> 1;   // 0..127
  const int seg = t & 1;    // 32-elem segment
  for (int i = 0; i < nk; ++i) {
    int kk = i << 6;
    {
      const uint4* src = reinterpret_cast<const uint4*>(A + (size_t)(blockM + lr) * K + kk + seg * 32);
      uint4* dst = reinterpret_cast<uint4*>(&Ash[0][lr * GLD + seg * 32]);
      dst[0] = src[0]; dst[1] = src[1]; dst[2] = src[2]; dst[3] = src[3];
    }
    {
      const uint4* src = reinterpret_cast<const uint4*>(Bt + (size_t)(blockN + lr) * K + kk + seg * 32);
      uint4* dst = reinterpret_cast<uint4*>(&Bsh[0][lr * GLD + seg * 32]);
      dst[0] = src[0]; dst[1] = src[1]; dst[2] = src[2]; dst[3] = src[3];
    }
    __syncthreads();
#pragma unroll
    for (int k2 = 0; k2 < 64; k2 += 32) {
      v16bf af[2];
#pragma unroll
      for (int mi = 0; mi < 2; ++mi) {
        const __bf16* ap = &Ash[0][(wm + mi * 16 + lm) * GLD + k2 + g * 8];
        af[mi] = cat16(*(const v8bf*)ap, *(const v8bf*)(ap + 16));
      }
      v16bf bfr[4];
#pragma unroll
      for (int ni = 0; ni < 4; ++ni) {
        const __bf16* bp = &Bsh[0][(wn + ni * 16 + lm) * GLD + k2 + g * 8];
        bfr[ni] = cat16(*(const v8bf*)bp, *(const v8bf*)(bp + 16));
      }
#pragma unroll
      for (int mi = 0; mi < 2; ++mi)
#pragma unroll
        for (int ni = 0; ni < 4; ++ni)
          acc[mi][ni] = __builtin_amdgcn_wmma_f32_16x16x32_bf16(
              false, af[mi], false, bfr[ni], (short)0, acc[mi][ni], false, false);
    }
    __syncthreads();
  }
#endif

  // epilogue: C layout row = 8*g + r, col = lane%16 within each 16x16 tile
#pragma unroll
  for (int mi = 0; mi < 2; ++mi)
#pragma unroll
    for (int ni = 0; ni < 4; ++ni)
#pragma unroll
      for (int r = 0; r < 8; ++r) {
        int row = blockM + wm + mi * 16 + 8 * g + r;
        int col = blockN + wn + ni * 16 + lm;
        float c = acc[mi][ni][r] + bias[col];
        size_t o = (size_t)row * N + col;
        if (EPI == EPI_F32) {
          ((float*)outp)[o] = c;
        } else if (EPI == EPI_BF16) {
          ((__bf16*)outp)[o] = (__bf16)c;
        } else if (EPI == EPI_RESID_F32) {
          ((float*)outp)[o] = c + resid[o];
        } else {  // exact GELU
          float gg = 0.5f * c * (1.f + erff(c * 0.70710678118f));
          ((__bf16*)outp)[o] = (__bf16)gg;
        }
      }
}

// ---------------------------------------------------------------------------
// Attention: one 512-thread block (16 waves) per (batch, head).
// NP=256, DK=64. Each wave owns 16 query rows: S strip 16x256 in registers,
// softmax via register + shfl_xor(width 16) reductions, P@V via WMMA with P
// staged through per-wave LDS and V transposed in LDS.
// q,k,v,attn: bf16 [B*NP, 512], head slab at col h*64.
// ---------------------------------------------------------------------------
__global__ __launch_bounds__(512)
void attention_kernel(const __bf16* __restrict__ q, const __bf16* __restrict__ k,
                      const __bf16* __restrict__ v, __bf16* __restrict__ attn)
{
  __shared__ __bf16 vt[64 * 264];           // V^T : vt[d][kk], padded stride 264
  __shared__ __bf16 psw[16 * 16 * PLDT];    // per-wave P staging: 16 rows x 32 cols

  const int t    = threadIdx.x;
  const int lane = t & 31;
  const int wid  = t >> 5;
  const int lm   = lane & 15;
  const int g    = lane >> 4;
  const int bh   = blockIdx.x;     // 0..511
  const int b    = bh >> 3;
  const int h    = bh & 7;
  const size_t rowbase = (size_t)b * 256;
  const int hoff = h * 64;

  // stage V transposed into LDS
  for (int i = t; i < 64 * 256; i += 512) {
    int d  = i >> 8;
    int kv = i & 255;
    vt[d * 264 + kv] = v[(rowbase + kv) * 512 + hoff + d];
  }

  // ---- S = q @ k^T (A,B fragments straight from global; rows are contiguous)
  v8f s[16];
#pragma unroll
  for (int ni = 0; ni < 16; ++ni) s[ni] = vzero8();

#pragma unroll
  for (int g2 = 0; g2 < 64; g2 += 32) {
    const __bf16* qp = &q[(rowbase + wid * 16 + lm) * 512 + hoff + g2 + g * 8];
    v16bf af = cat16(*(const v8bf*)qp, *(const v8bf*)(qp + 16));
#pragma unroll
    for (int ni = 0; ni < 16; ++ni) {
      const __bf16* kp = &k[(rowbase + ni * 16 + lm) * 512 + hoff + g2 + g * 8];
      v16bf bfr = cat16(*(const v8bf*)kp, *(const v8bf*)(kp + 16));
      s[ni] = __builtin_amdgcn_wmma_f32_16x16x32_bf16(
          false, af, false, bfr, (short)0, s[ni], false, false);
    }
  }

  // ---- softmax over 256 cols; lane holds rows (8*g + r), cols (ni*16 + lm)
  float rmax[8], rinv[8];
#pragma unroll
  for (int r = 0; r < 8; ++r) {
    float m = -1e30f;
#pragma unroll
    for (int ni = 0; ni < 16; ++ni) m = fmaxf(m, s[ni][r]);
#pragma unroll
    for (int off = 1; off < 16; off <<= 1) m = fmaxf(m, __shfl_xor(m, off, 32));
    rmax[r] = m;
  }
#pragma unroll
  for (int r = 0; r < 8; ++r) {
    float sum = 0.f;
#pragma unroll
    for (int ni = 0; ni < 16; ++ni) {
      float e = __expf(0.125f * (s[ni][r] - rmax[r]));  // inv_scale = 1/sqrt(64)
      s[ni][r] = e;
      sum += e;
    }
#pragma unroll
    for (int off = 1; off < 16; off <<= 1) sum += __shfl_xor(sum, off, 32);
    rinv[r] = 1.f / sum;
  }

  __syncthreads();  // vt ready for all waves

  // ---- O = P @ V ; stage P (C-layout -> A-layout) through per-wave LDS
  v8f o[4];
#pragma unroll
  for (int nd = 0; nd < 4; ++nd) o[nd] = vzero8();

  __bf16* myp = &psw[wid * 16 * PLDT];
#pragma unroll
  for (int kb = 0; kb < 8; ++kb) {
#pragma unroll
    for (int nl = 0; nl < 2; ++nl) {
      int ni = kb * 2 + nl;
#pragma unroll
      for (int r = 0; r < 8; ++r)
        myp[(8 * g + r) * PLDT + nl * 16 + lm] = (__bf16)(s[ni][r] * rinv[r]);
    }
    // same-wave LDS ordering is guaranteed (DS ops in order); build A fragment
    const __bf16* ap = &myp[lm * PLDT + g * 8];
    v16bf af = cat16(*(const v8bf*)ap, *(const v8bf*)(ap + 16));
#pragma unroll
    for (int nd = 0; nd < 4; ++nd) {
      const __bf16* bp = &vt[(nd * 16 + lm) * 264 + kb * 32 + g * 8];
      v16bf bfr = cat16(*(const v8bf*)bp, *(const v8bf*)(bp + 16));
      o[nd] = __builtin_amdgcn_wmma_f32_16x16x32_bf16(
          false, af, false, bfr, (short)0, o[nd], false, false);
    }
  }

#pragma unroll
  for (int nd = 0; nd < 4; ++nd)
#pragma unroll
    for (int r = 0; r < 8; ++r) {
      int grow = wid * 16 + 8 * g + r;
      int col  = nd * 16 + lm;
      attn[(rowbase + grow) * 512 + hoff + col] = (__bf16)o[nd][r];
    }
}

// ---------------------------------------------------------------------------
// LayerNorm over D=512 rows. One 256-thread block per row (2 elems/thread).
// Modes: 0 -> bf16 out, 1 -> f32 out, 2 -> f32 out + pos_enc[row%256]
// ---------------------------------------------------------------------------
enum { LNO_BF16 = 0, LNO_F32 = 1, LNO_F32_POS = 2 };

template<int MODE>
__global__ __launch_bounds__(256)
void ln_kernel(const float* __restrict__ in, const float* __restrict__ w,
               const float* __restrict__ bb, const float* __restrict__ pos,
               void* __restrict__ outp)
{
  __shared__ float red[256];
  const int row = blockIdx.x;
  const int t = threadIdx.x;
  const float* x = in + (size_t)row * 512;
  float v0 = x[t], v1 = x[t + 256];

  red[t] = v0 + v1;
  __syncthreads();
  for (int off = 128; off > 0; off >>= 1) {
    if (t < off) red[t] += red[t + off];
    __syncthreads();
  }
  float mean = red[0] * (1.f / 512.f);
  __syncthreads();

  float d0 = v0 - mean, d1 = v1 - mean;
  red[t] = d0 * d0 + d1 * d1;
  __syncthreads();
  for (int off = 128; off > 0; off >>= 1) {
    if (t < off) red[t] += red[t + off];
    __syncthreads();
  }
  float rstd = rsqrtf(red[0] * (1.f / 512.f) + 1e-5f);

  float y0 = d0 * rstd * w[t] + bb[t];
  float y1 = d1 * rstd * w[t + 256] + bb[t + 256];
  size_t o0 = (size_t)row * 512 + t;
  size_t o1 = o0 + 256;
  if (MODE == LNO_BF16) {
    __bf16* o = (__bf16*)outp;
    o[o0] = (__bf16)y0; o[o1] = (__bf16)y1;
  } else if (MODE == LNO_F32) {
    float* o = (float*)outp;
    o[o0] = y0; o[o1] = y1;
  } else {
    int n = row & 255;  // NP = 256
    float* o = (float*)outp;
    o[o0] = y0 + pos[n * 512 + t];
    o[o1] = y1 + pos[n * 512 + t + 256];
  }
}

// f32 W[K][N] -> bf16 Wt[N][K]
__global__ __launch_bounds__(256)
void wcast_transpose(const float* __restrict__ w, __bf16* __restrict__ wt, int K, int N)
{
  int idx = blockIdx.x * 256 + threadIdx.x;
  if (idx >= K * N) return;
  int kq = idx / N, n = idx % N;
  wt[(size_t)n * K + kq] = (__bf16)w[idx];
}

// Effective QKV weight with LoRA folded: Wt[n][k] = bf16(w[k][n] + 2*sum_r down[k][r]*up[r][n])
__global__ __launch_bounds__(256)
void qkv_eff(const float* __restrict__ w, const float* __restrict__ down,
             const float* __restrict__ up, __bf16* __restrict__ wt)
{
  int idx = blockIdx.x * 256 + threadIdx.x;
  if (idx >= 512 * 512) return;
  int kq = idx >> 9, n = idx & 511;
  float acc = w[idx];
#pragma unroll
  for (int r = 0; r < 8; ++r) acc += 2.0f * down[kq * 8 + r] * up[r * 512 + n];
  wt[(size_t)n * 512 + kq] = (__bf16)acc;
}

// patches[b*256+n][f*16+t] = bf16(x[b][n*16+t][f])
__global__ __launch_bounds__(256)
void build_patches(const float* __restrict__ x, __bf16* __restrict__ pt)
{
  int idx = blockIdx.x * 256 + threadIdx.x;
  if (idx >= 16384 * 128) return;
  int m = idx >> 7, c = idx & 127;
  int f = c >> 4, tt = c & 15;
  int b = m >> 8, n = m & 255;
  pt[idx] = (__bf16)x[((size_t)b * 4096 + n * 16 + tt) * 8 + f];
}

// out[b][d] = mean over 256 patches
__global__ __launch_bounds__(256)
void mean_kernel(const float* __restrict__ tmp, float* __restrict__ out)
{
  int idx = blockIdx.x * 256 + threadIdx.x;
  if (idx >= 64 * 512) return;
  int b = idx >> 9, dd = idx & 511;
  float sacc = 0.f;
  for (int n = 0; n < 256; ++n) sacc += tmp[((size_t)b * 256 + n) * 512 + dd];
  out[idx] = sacc * (1.f / 256.f);
}

// ---------------------------------------------------------------------------
extern "C" void kernel_launch(void* const* d_in, const int* in_sizes, int n_in,
                              void* d_out, int out_size, void* d_ws, size_t ws_size,
                              hipStream_t stream)
{
  const float* x         = (const float*)d_in[0];
  const float* patch_w   = (const float*)d_in[1];
  const float* patch_b   = (const float*)d_in[2];
  const float* patch_lnw = (const float*)d_in[3];
  const float* patch_lnb = (const float*)d_in[4];
  const float* pos_enc   = (const float*)d_in[5];
  const float* ln1_w = (const float*)d_in[6];
  const float* ln1_b = (const float*)d_in[7];
  const float* q_w  = (const float*)d_in[8];
  const float* q_b  = (const float*)d_in[9];
  const float* k_w  = (const float*)d_in[10];
  const float* k_b  = (const float*)d_in[11];
  const float* v_w  = (const float*)d_in[12];
  const float* v_b  = (const float*)d_in[13];
  const float* lq_d = (const float*)d_in[14];
  const float* lq_u = (const float*)d_in[15];
  const float* lk_d = (const float*)d_in[16];
  const float* lk_u = (const float*)d_in[17];
  const float* lv_d = (const float*)d_in[18];
  const float* lv_u = (const float*)d_in[19];
  const float* o_w  = (const float*)d_in[20];
  const float* o_b  = (const float*)d_in[21];
  const float* ln2_w = (const float*)d_in[22];
  const float* ln2_b = (const float*)d_in[23];
  const float* ff1_w = (const float*)d_in[24];
  const float* ff1_b = (const float*)d_in[25];
  const float* ff2_w = (const float*)d_in[26];
  const float* ff2_b = (const float*)d_in[27];
  const float* norm_w = (const float*)d_in[28];
  const float* norm_b = (const float*)d_in[29];

  const int M = 16384;  // B*NP
  char* p = (char*)d_ws;
  auto alloc = [&](size_t bytes) -> void* {
    void* r = (void*)p;
    p += (bytes + 255) & ~(size_t)255;
    return r;
  };

  __bf16* patches = (__bf16*)alloc((size_t)M * 128 * 2);
  float*  hbuf    = (float*) alloc((size_t)M * 512 * 4);
  float*  tmp     = (float*) alloc((size_t)M * 512 * 4);
  __bf16* yln     = (__bf16*)alloc((size_t)M * 512 * 2);
  __bf16* qb      = (__bf16*)alloc((size_t)M * 512 * 2);
  __bf16* kbuf    = (__bf16*)alloc((size_t)M * 512 * 2);
  __bf16* vbuf    = (__bf16*)alloc((size_t)M * 512 * 2);
  __bf16* attnb   = (__bf16*)alloc((size_t)M * 512 * 2);
  __bf16* ffmid   = (__bf16*)alloc((size_t)M * 2048 * 2);
  __bf16* wpt     = (__bf16*)alloc((size_t)512 * 128 * 2);
  __bf16* wq      = (__bf16*)alloc((size_t)512 * 512 * 2);
  __bf16* wk      = (__bf16*)alloc((size_t)512 * 512 * 2);
  __bf16* wv      = (__bf16*)alloc((size_t)512 * 512 * 2);
  __bf16* wo      = (__bf16*)alloc((size_t)512 * 512 * 2);
  __bf16* wff1    = (__bf16*)alloc((size_t)512 * 2048 * 2);
  __bf16* wff2    = (__bf16*)alloc((size_t)2048 * 512 * 2);

  dim3 blk(256);

  // patch embedding: gather -> GEMM -> LN + pos
  build_patches<<<(M * 128 + 255) / 256, blk, 0, stream>>>(x, patches);
  wcast_transpose<<<(128 * 512 + 255) / 256, blk, 0, stream>>>(patch_w, wpt, 128, 512);
  gemm_bf16<EPI_F32><<<dim3(4, 128), blk, 0, stream>>>(patches, wpt, patch_b, nullptr, tmp, M, 512, 128);
  ln_kernel<LNO_F32_POS><<<M, blk, 0, stream>>>(tmp, patch_lnw, patch_lnb, pos_enc, hbuf);

  for (int i = 0; i < 6; ++i) {
    size_t dd = (size_t)i * 512 * 512;
    size_t dr = (size_t)i * 512 * 8;
    size_t d1 = (size_t)i * 512;
    size_t dF = (size_t)i * 512 * 2048;
    size_t dFb = (size_t)i * 2048;

    // per-layer weight prep (bf16, N-major, LoRA folded into QKV)
    qkv_eff<<<(512 * 512 + 255) / 256, blk, 0, stream>>>(q_w + dd, lq_d + dr, lq_u + dr, wq);
    qkv_eff<<<(512 * 512 + 255) / 256, blk, 0, stream>>>(k_w + dd, lk_d + dr, lk_u + dr, wk);
    qkv_eff<<<(512 * 512 + 255) / 256, blk, 0, stream>>>(v_w + dd, lv_d + dr, lv_u + dr, wv);
    wcast_transpose<<<(512 * 512 + 255) / 256, blk, 0, stream>>>(o_w + dd, wo, 512, 512);
    wcast_transpose<<<(512 * 2048 + 255) / 256, blk, 0, stream>>>(ff1_w + dF, wff1, 512, 2048);
    wcast_transpose<<<(2048 * 512 + 255) / 256, blk, 0, stream>>>(ff2_w + dF, wff2, 2048, 512);

    // attention sublayer
    ln_kernel<LNO_BF16><<<M, blk, 0, stream>>>(hbuf, ln1_w + d1, ln1_b + d1, nullptr, yln);
    gemm_bf16<EPI_BF16><<<dim3(4, 128), blk, 0, stream>>>(yln, wq, q_b + d1, nullptr, qb,   M, 512, 512);
    gemm_bf16<EPI_BF16><<<dim3(4, 128), blk, 0, stream>>>(yln, wk, k_b + d1, nullptr, kbuf, M, 512, 512);
    gemm_bf16<EPI_BF16><<<dim3(4, 128), blk, 0, stream>>>(yln, wv, v_b + d1, nullptr, vbuf, M, 512, 512);
    attention_kernel<<<512, dim3(512), 0, stream>>>(qb, kbuf, vbuf, attnb);
    gemm_bf16<EPI_RESID_F32><<<dim3(4, 128), blk, 0, stream>>>(attnb, wo, o_b + d1, hbuf, hbuf, M, 512, 512);

    // FFN sublayer
    ln_kernel<LNO_BF16><<<M, blk, 0, stream>>>(hbuf, ln2_w + d1, ln2_b + d1, nullptr, yln);
    gemm_bf16<EPI_GELU_BF16><<<dim3(16, 128), blk, 0, stream>>>(yln, wff1, ff1_b + dFb, nullptr, ffmid, M, 2048, 512);
    gemm_bf16<EPI_RESID_F32><<<dim3(4, 128), blk, 0, stream>>>(ffmid, wff2, ff2_b + d1, hbuf, hbuf, M, 512, 2048);
  }

  ln_kernel<LNO_F32><<<M, blk, 0, stream>>>(hbuf, norm_w, norm_b, nullptr, tmp);
  mean_kernel<<<(64 * 512 + 255) / 256, blk, 0, stream>>>(tmp, (float*)d_out);
}
